// MLA_16501264351753
// MI455X (gfx1250) — matrix operand order repaired and checked by reading, not compiled, not run on hardware
//
#include <hip/hip_runtime.h>

typedef _Float16 h16;
typedef __attribute__((ext_vector_type(8)))  _Float16 v8h;
typedef __attribute__((ext_vector_type(16))) _Float16 v16h;
typedef __attribute__((ext_vector_type(8)))  float    v8f;

#define B_  2
#define S_  2048
#define D_  2048
#define H_  16
#define HD_ 128
#define L_  512
#define R_  64
#define RH_ 32
#define M_  (B_*S_)                      // 4096 token rows
#define SCALE_ 0.08838834764831845f      // 128^-0.5
#define NEG_ (-1.0e9f)

// ---------------------------------------------------------------- WMMA helpers
__device__ inline v8f wmma16x16x32(v16h a, v16h b, v8f c) {
  // (neg_a, A, neg_b, B, c_mod, C, reuse_a, reuse_b)
  return __builtin_amdgcn_wmma_f32_16x16x32_f16(false, a, false, b, (short)0, c,
                                                false, false);
}
__device__ inline v16h cat8(v8h lo, v8h hi) {
  return __builtin_shufflevector(lo, hi, 0,1,2,3,4,5,6,7,8,9,10,11,12,13,14,15);
}

// ---------------------------------------------------------------- f32 -> f16
__global__ void cvt_f32_f16(const float* __restrict__ in, h16* __restrict__ out,
                            long n) {
  long i = (long)blockIdx.x * blockDim.x + threadIdx.x;
  long stride = (long)gridDim.x * blockDim.x;
  for (; i < n; i += stride) out[i] = (h16)in[i];
}

// ---------------------------------------------------------------- tiled GEMM
// C[M,N] = A[M,K] @ B[K,N], A/B f16 row-major, OutT in {f16, f32}.
// Block: 256 threads (8 wave32), block tile 128x64, wave tile 32x32.
// A staged in LDS [128][32]; B staged transposed [64][32] so that both
// fragment reads are contiguous ds_load_b128 matching WMMA 16-bit layouts.
template <typename OutT>
__global__ void __launch_bounds__(256) gemm_wmma(
    const h16* __restrict__ A, const h16* __restrict__ B, OutT* __restrict__ C,
    int Mdim, int Ndim, int Kdim) {
  __shared__ h16 As[128 * 32];
  __shared__ h16 BsT[64 * 32];
  const int t    = threadIdx.x;
  const int lane = t & 31;
  const int w    = t >> 5;
  const int wm   = w & 3;        // 4 waves along M
  const int wn   = w >> 2;       // 2 waves along N
  const int nl   = lane & 15;
  const int hi   = lane >> 4;    // half-wave select
  const int mblock = blockIdx.y * 128;
  const int nblock = blockIdx.x * 64;

  v8f z = {0,0,0,0,0,0,0,0};
  v8f acc[2][2] = {{z,z},{z,z}};

  const int arow  = t >> 1;        // 0..127
  const int ahalf = (t & 1) * 16;  // 0 or 16
  const int brow  = t >> 3;        // 0..31
  const int bcol  = (t & 7) * 8;   // 0..56

  const h16* aptr = A + (size_t)(mblock + arow) * Kdim + ahalf;
  const h16* bptr = B + (size_t)brow * Ndim + nblock + bcol;

  for (int kb = 0; kb < Kdim; kb += 32) {
    __syncthreads();
    { // stage A tile 128x32 (two b128 per thread, coalesced)
      const h16* src = aptr + kb;
      *(v8h*)&As[arow * 32 + ahalf]     = *(const v8h*)src;
      *(v8h*)&As[arow * 32 + ahalf + 8] = *(const v8h*)(src + 8);
    }
    { // stage B tile 32x64, transposed into BsT[n][k]
      v8h bv = *(const v8h*)(bptr + (size_t)kb * Ndim);
      #pragma unroll
      for (int i = 0; i < 8; ++i) BsT[(bcol + i) * 32 + brow] = bv[i];
    }
    if (kb + 32 < Kdim) {  // global_prefetch_b8 of next K tile
      __builtin_prefetch(aptr + kb + 32, 0, 1);
      __builtin_prefetch(bptr + (size_t)(kb + 32) * Ndim, 0, 1);
    }
    __syncthreads();

    v16h af[2], bf[2];
    #pragma unroll
    for (int mt = 0; mt < 2; ++mt) {  // A frag: lanes 0-15 K0..7|16..23, hi half +8
      int row = wm * 32 + mt * 16 + nl;
      v8h lo = *(v8h*)&As[row * 32 + hi * 8];
      v8h hh = *(v8h*)&As[row * 32 + 16 + hi * 8];
      af[mt] = cat8(lo, hh);
    }
    #pragma unroll
    for (int nt = 0; nt < 2; ++nt) {  // B frag: lanes 0-15 K0..7|16..23, hi half +8
      int col = wn * 32 + nt * 16 + nl;
      v8h lo = *(v8h*)&BsT[col * 32 + hi * 8];
      v8h hh = *(v8h*)&BsT[col * 32 + 16 + hi * 8];
      bf[nt] = cat8(lo, hh);
    }
    #pragma unroll
    for (int mt = 0; mt < 2; ++mt)
      #pragma unroll
      for (int nt = 0; nt < 2; ++nt)
        acc[mt][nt] = wmma16x16x32(af[mt], bf[nt], acc[mt][nt]);
  }

  // C layout: VGPR r -> M = r + 8*hi, N = nl
  #pragma unroll
  for (int mt = 0; mt < 2; ++mt)
    #pragma unroll
    for (int nt = 0; nt < 2; ++nt) {
      int n = nblock + wn * 32 + nt * 16 + nl;
      #pragma unroll
      for (int r = 0; r < 8; ++r) {
        int m = mblock + wm * 32 + mt * 16 + r + 8 * hi;
        C[(size_t)m * Ndim + n] = (OutT)acc[mt][nt][r];
      }
    }
}

// ---------------------------------------------------------------- RoPE in-place
// p: [nrows, R]; position = (row / posdiv) % S
__global__ void rope_inplace(h16* __restrict__ p, int nrows, int posdiv) {
  int tid = blockIdx.x * blockDim.x + threadIdx.x;
  if (tid >= nrows * RH_) return;
  int row = tid / RH_;
  int i   = tid - row * RH_;
  int pos = (row / posdiv) % S_;
  float inv = __powf(10000.0f, -(float)(2 * i) / (float)R_);
  float f = (float)pos * inv;
  float sn, cs;
  __sincosf(f, &sn, &cs);
  size_t base = (size_t)row * R_ + 2 * i;
  float xe = (float)p[base];
  float xo = (float)p[base + 1];
  p[base]     = (h16)(xe * cs - xo * sn);
  p[base + 1] = (h16)(xe * sn + xo * cs);
}

// ---------------------------------------------------------------- flash attention
// One wave per (b, h, 16-row q tile). d_qk = 192 (q_c||q_r), d_v = 128.
// K-fragment loads are software-pipelined (2-deep double buffer); V-tile global
// loads are issued into a register array right after the top barrier and only
// drained to LDS after softmax, so their latency is hidden under the 12 score
// WMMAs + softmax instead of serializing load->ds_store 16 times.
__global__ void __launch_bounds__(32) mla_attn(
    const h16* __restrict__ kv,   // [M, 2D]  (k_c cols [h*128..], v cols [2048 + h*128..])
    const h16* __restrict__ qc,   // [M, D]
    const h16* __restrict__ qr,   // [M, H*R]
    const h16* __restrict__ kr,   // [M, R]
    h16* __restrict__ att) {      // [M, D]  == (B,S,H,HD)
  __shared__ h16 Pt[16 * 32];     // P tile for A-layout transpose
  __shared__ h16 Vs[32 * 128];    // staged V tile
  const int l  = threadIdx.x;
  const int nl = l & 15;
  const int hi = l >> 4;

  const int QT = S_ / 16;
  int tile = blockIdx.x;
  int q0 = (tile % QT) * 16;
  int h  = (tile / QT) % H_;
  int b  = tile / (QT * H_);

  // Q fragments: 6 chunks of 32 along d (chunks 0-3 from q_c, 4-5 from q_r)
  v16h qf[6];
  {
    size_t bs = (size_t)(b * S_ + q0 + nl);
    const h16* qcrow = qc + bs * (size_t)D_ + h * HD_;
    const h16* qrrow = qr + bs * (size_t)(H_ * R_) + h * R_;
    #pragma unroll
    for (int c = 0; c < 6; ++c) {
      const h16* base = (c < 4) ? (qcrow + 32 * c) : (qrrow + 32 * (c - 4));
      v8h lo = *(const v8h*)(base + 16 * hi);
      v8h hh = *(const v8h*)(base + 16 * hi + 8);
      qf[c] = cat8(lo, hh);
    }
  }

  // Hoisted per-lane base pointers (advance by flat offsets per key tile).
  const h16* kcp[2];  // into kv (k_c part), includes +16*hi d-offset
  const h16* krp[2];  // into kr,            includes +16*hi d-offset
  #pragma unroll
  for (int nt = 0; nt < 2; ++nt) {
    size_t row0 = (size_t)(b * S_ + nt * 16 + nl);
    kcp[nt] = kv + row0 * (size_t)(2 * D_) + h * HD_ + 16 * hi;
    krp[nt] = kr + row0 * (size_t)R_ + 16 * hi;
  }
  const h16* vsrc0 = kv + (size_t)(b * S_) * (size_t)(2 * D_) + D_ + h * HD_;

  v8f z = {0,0,0,0,0,0,0,0};
  v8f o[8];
  #pragma unroll
  for (int t = 0; t < 8; ++t) o[t] = z;
  float m_i[8], l_i[8];
  #pragma unroll
  for (int r = 0; r < 8; ++r) { m_i[r] = -3.0e38f; l_i[r] = 0.0f; }

  int ktmax = (q0 + 15) / 32;  // causal: last 32-key tile touching this q tile
  for (int kt = 0; kt <= ktmax; ++kt) {
    const size_t kcoff = (size_t)kt * 32 * (size_t)(2 * D_);
    const size_t kroff = (size_t)kt * 32 * (size_t)R_;

    __syncthreads();  // prev iteration's Pt/Vs reads complete before new writes

    // Issue V tile loads (32 keys x 128 dims) into registers; drain later.
    v8h vtmp[16];
    #pragma unroll
    for (int c = 0; c < 16; ++c) {
      int id  = c * 32 + l;
      int row = id >> 4;
      int off = (id & 15) * 8;
      vtmp[c] = *(const v8h*)(vsrc0 + kcoff + (size_t)row * (size_t)(2 * D_) + off);
    }

    // scores: two 16x16 key sub-tiles, 6 d-chunks, double-buffered K frags
    v8f s0 = z, s1 = z;
    v16h kbuf[2][2];
    #pragma unroll
    for (int nt = 0; nt < 2; ++nt) {     // prologue: chunk 0
      const h16* base = kcp[nt] + kcoff;
      kbuf[0][nt] = cat8(*(const v8h*)base, *(const v8h*)(base + 8));
    }
    #pragma unroll
    for (int c = 0; c < 6; ++c) {
      if (c < 5) {  // issue chunk c+1 loads before consuming chunk c
        int cn = c + 1;
        #pragma unroll
        for (int nt = 0; nt < 2; ++nt) {
          const h16* base = (cn < 4) ? (kcp[nt] + kcoff + 32 * cn)
                                     : (krp[nt] + kroff + 32 * (cn - 4));
          kbuf[(c + 1) & 1][nt] =
              cat8(*(const v8h*)base, *(const v8h*)(base + 8));
        }
      }
      s0 = wmma16x16x32(qf[c], kbuf[c & 1][0], s0);
      s1 = wmma16x16x32(qf[c], kbuf[c & 1][1], s1);
    }

    // online softmax (rows live in 16-lane groups; xor masks < 16 stay in-group)
    #pragma unroll
    for (int r = 0; r < 8; ++r) {
      int rowg = q0 + r + 8 * hi;
      float x0 = s0[r] * SCALE_;
      float x1 = s1[r] * SCALE_;
      if (kt * 32 + nl      > rowg) x0 = NEG_;
      if (kt * 32 + 16 + nl > rowg) x1 = NEG_;
      float mv = fmaxf(x0, x1);
      #pragma unroll
      for (int d = 1; d < 16; d <<= 1) mv = fmaxf(mv, __shfl_xor(mv, d, 32));
      float mnew = fmaxf(m_i[r], mv);
      float p0 = __expf(x0 - mnew);
      float p1 = __expf(x1 - mnew);
      float srow = p0 + p1;
      #pragma unroll
      for (int d = 1; d < 16; d <<= 1) srow += __shfl_xor(srow, d, 32);
      float alpha = __expf(m_i[r] - mnew);
      l_i[r] = l_i[r] * alpha + srow;
      m_i[r] = mnew;
      #pragma unroll
      for (int t = 0; t < 8; ++t) o[t][r] *= alpha;
      Pt[(r + 8 * hi) * 32 + nl]      = (h16)p0;
      Pt[(r + 8 * hi) * 32 + 16 + nl] = (h16)p1;
    }

    // drain V tile to LDS (loads have had the whole score phase to land)
    #pragma unroll
    for (int c = 0; c < 16; ++c) {
      int id  = c * 32 + l;
      int row = id >> 4;
      int off = (id & 15) * 8;
      *(v8h*)&Vs[row * 128 + off] = vtmp[c];
    }
    __syncthreads();

    // P(16x32) @ V(32x128): P re-read in WMMA A layout, V gathered from LDS
    v16h ap;
    {
      v8h lo = *(v8h*)&Pt[nl * 32 + 16 * hi];
      v8h hh = *(v8h*)&Pt[nl * 32 + 16 * hi + 8];
      ap = cat8(lo, hh);
    }
    #pragma unroll
    for (int t = 0; t < 8; ++t) {
      v16h bv;
      #pragma unroll
      for (int i = 0; i < 16; ++i)
        bv[i] = Vs[(16 * hi + i) * 128 + t * 16 + nl];
      o[t] = wmma16x16x32(ap, bv, o[t]);
    }
  }
  // normalize + store attended in (B,S,H,HD) layout == [M, D]
  #pragma unroll
  for (int r = 0; r < 8; ++r) {
    float inv = 1.0f / l_i[r];
    int mg = q0 + r + 8 * hi;
    #pragma unroll
    for (int t = 0; t < 8; ++t)
      att[(size_t)(b * S_ + mg) * (size_t)D_ + h * HD_ + t * 16 + nl] =
          (h16)(o[t][r] * inv);
  }
}

// ---------------------------------------------------------------- launcher
extern "C" void kernel_launch(void* const* d_in, const int* in_sizes, int n_in,
                              void* d_out, int out_size, void* d_ws, size_t ws_size,
                              hipStream_t stream) {
  (void)in_sizes; (void)n_in; (void)out_size; (void)ws_size;
  const float* x    = (const float*)d_in[0];
  const float* Wckv = (const float*)d_in[1];
  const float* Wkv  = (const float*)d_in[2];
  const float* Wcq  = (const float*)d_in[3];
  const float* Wq   = (const float*)d_in[4];
  const float* Wqr  = (const float*)d_in[5];
  const float* Wkr  = (const float*)d_in[6];
  const float* Wout = (const float*)d_in[7];
  float* out = (float*)d_out;

  char* ws = (char*)d_ws;
  auto alloc = [&](size_t halves) -> h16* {
    h16* p = (h16*)ws;
    ws += ((halves * sizeof(h16) + 255) & ~(size_t)255);
    return p;
  };
  h16* xb   = alloc((size_t)M_ * D_);
  h16* wckv = alloc((size_t)D_ * L_);
  h16* wkv  = alloc((size_t)L_ * 2 * D_);
  h16* wcq  = alloc((size_t)D_ * L_);
  h16* wq   = alloc((size_t)L_ * D_);
  h16* wqr  = alloc((size_t)L_ * H_ * R_);
  h16* wkr  = alloc((size_t)L_ * R_);
  h16* wout = alloc((size_t)D_ * D_);
  h16* ckv  = alloc((size_t)M_ * L_);
  h16* cq   = alloc((size_t)M_ * L_);
  h16* kvm  = alloc((size_t)M_ * 2 * D_);
  h16* qcm  = alloc((size_t)M_ * D_);
  h16* qrm  = alloc((size_t)M_ * H_ * R_);
  h16* krm  = alloc((size_t)M_ * R_);
  h16* att  = alloc((size_t)M_ * D_);

  auto cvt = [&](const float* src, h16* dst, long n) {
    cvt_f32_f16<<<1024, 256, 0, stream>>>(src, dst, n);
  };
  cvt(x,    xb,   (long)M_ * D_);
  cvt(Wckv, wckv, (long)D_ * L_);
  cvt(Wkv,  wkv,  (long)L_ * 2 * D_);
  cvt(Wcq,  wcq,  (long)D_ * L_);
  cvt(Wq,   wq,   (long)L_ * D_);
  cvt(Wqr,  wqr,  (long)L_ * H_ * R_);
  cvt(Wkr,  wkr,  (long)L_ * R_);
  cvt(Wout, wout, (long)D_ * D_);

  auto gemm16 = [&](const h16* A, const h16* Bm, h16* Cm, int Mm, int Nn, int Kk) {
    dim3 g(Nn / 64, Mm / 128);
    gemm_wmma<h16><<<g, 256, 0, stream>>>(A, Bm, Cm, Mm, Nn, Kk);
  };
  gemm16(xb,  wckv, ckv, M_, L_,       D_);   // c_kv
  gemm16(xb,  wcq,  cq,  M_, L_,       D_);   // c_q
  gemm16(ckv, wkv,  kvm, M_, 2 * D_,   L_);   // kv = [k_c | v]
  gemm16(cq,  wq,   qcm, M_, D_,       L_);   // q_c
  gemm16(cq,  wqr,  qrm, M_, H_ * R_,  L_);   // q_r
  gemm16(ckv, wkr,  krm, M_, R_,       L_);   // k_r

  {
    int nq = M_ * H_ * RH_;
    rope_inplace<<<(nq + 255) / 256, 256, 0, stream>>>(qrm, M_ * H_, H_);
    int nk = M_ * RH_;
    rope_inplace<<<(nk + 255) / 256, 256, 0, stream>>>(krm, M_, 1);
  }

  mla_attn<<<B_ * H_ * (S_ / 16), 32, 0, stream>>>(kvm, qcm, qrm, krm, att);

  {
    dim3 g(D_ / 64, M_ / 128);
    gemm_wmma<float><<<g, 256, 0, stream>>>(att, wout, out, M_, D_, D_);
  }
}
